// Attention_34763465294455
// MI455X (gfx1250) — matrix operand order repaired and checked by reading, not compiled, not run on hardware
//
#include <hip/hip_runtime.h>
#include <hip/hip_bf16.h>

// ---------- CDNA5 (gfx1250) types ----------
typedef __attribute__((ext_vector_type(16))) __bf16 bf16x16;
typedef __attribute__((ext_vector_type(8)))  float  f32x8;

union FragU {
  bf16x16 v;
  uint4   q[2];
};

__device__ __forceinline__ __bf16 f2bf(float f) {
  unsigned u = __builtin_bit_cast(unsigned, f);
  u += 0x7fffu + ((u >> 16) & 1u);                 // round-to-nearest-even
  unsigned short h = (unsigned short)(u >> 16);
  return __builtin_bit_cast(__bf16, h);
}

__device__ __forceinline__ float redmax16(float v) {
  v = fmaxf(v, __shfl_xor(v, 1, 16));
  v = fmaxf(v, __shfl_xor(v, 2, 16));
  v = fmaxf(v, __shfl_xor(v, 4, 16));
  v = fmaxf(v, __shfl_xor(v, 8, 16));
  return v;
}
__device__ __forceinline__ float redsum16(float v) {
  v += __shfl_xor(v, 1, 16);
  v += __shfl_xor(v, 2, 16);
  v += __shfl_xor(v, 4, 16);
  v += __shfl_xor(v, 8, 16);
  return v;
}

// ---------- CDNA5 async global->LDS DMA (ASYNCcnt-tracked, §15.18.3) ----------
// Per-lane 16B copy directly into LDS, bypassing VGPRs. The low 32 bits of a
// generic pointer to __shared__ are the AS3 (wave-relative) LDS offset.
__device__ __forceinline__ void async_g2l_b128(const void* lds, const void* g) {
  asm volatile("global_load_async_to_lds_b128 %0, %1, off"
               :: "v"((unsigned)(uintptr_t)lds), "v"((unsigned long long)(uintptr_t)g)
               : "memory");
}
__device__ __forceinline__ void wait_async0() {
  asm volatile("s_wait_asynccnt 0" ::: "memory");
}

// ---------- problem constants ----------
constexpr int CB = 2, CL = 2048, CD = 1024, CH = 16, CHD = 64;
constexpr int CM = CB * CL;          // 4096 rows

// ---------- fp32 -> bf16 conversion ----------
__global__ void cvt_bf16(const float* __restrict__ in, __bf16* __restrict__ out, int n) {
  union { __bf16 h[4]; uint2 u; } pk;
  int i = (blockIdx.x * blockDim.x + threadIdx.x) * 4;
  const int stride = gridDim.x * blockDim.x * 4;
  for (; i < n; i += stride) {
    float4 f = *(const float4*)(in + i);
    pk.h[0] = f2bf(f.x); pk.h[1] = f2bf(f.y); pk.h[2] = f2bf(f.z); pk.h[3] = f2bf(f.w);
    *(uint2*)(out + i) = pk.u;
  }
}

// ---------- tiled bf16 WMMA GEMM mainloop (shared by both projections) ----------
// Block tile 128(M) x 128(N) x 32(K); 8 waves as 2(M) x 4(N); wave tile 64x32.
// Double-buffered LDS; tile i+1 streams in via async DMA while tile i feeds
// the WMMA pipe; tile i+2 is prefetched into L2.
constexpr int LDT = 40;   // padded LDS stride (elements): 80B -> bank-conflict free

#define GEMM_MAINLOOP(Aptr, Wptr, Kdim)                                              \
  __shared__ __align__(16) __bf16 As[2][128 * LDT];                                  \
  __shared__ __align__(16) __bf16 Bs[2][128 * LDT];                                  \
  const int tid  = threadIdx.x;                                                      \
  const int wave = tid >> 5, lane = tid & 31;                                        \
  const int wm = wave >> 2, wn = wave & 3;                                           \
  const int lrow = lane & 15;                                                        \
  const int lhi  = lane >> 4;                                                        \
  const int lkA  = lhi << 3;      /* A operand: K split {0..7,16..23} */             \
  const int lkB  = lhi << 3;      /* B operand: K split {0..7,16..23} */             \
  const int m0 = blockIdx.y * 128, n0 = blockIdx.x * 128;                            \
  const int srow = tid >> 1;                                                         \
  const int scol = (tid & 1) << 4;                                                   \
  const __bf16* gA = (Aptr) + (size_t)(m0 + srow) * (Kdim) + scol;                   \
  const __bf16* gB = (Wptr) + (size_t)(n0 + srow) * (Kdim) + scol;                   \
  f32x8 acc[4][2] = {};                                                              \
  constexpr int NIT = (Kdim) / 32;                                                   \
  async_g2l_b128(&As[0][srow * LDT + scol],     gA);                                 \
  async_g2l_b128(&As[0][srow * LDT + scol + 8], gA + 8);                             \
  async_g2l_b128(&Bs[0][srow * LDT + scol],     gB);                                 \
  async_g2l_b128(&Bs[0][srow * LDT + scol + 8], gB + 8);                             \
  wait_async0();                                                                     \
  __syncthreads();                                                                   \
  for (int it = 0; it < NIT; ++it) {                                                 \
    const int cur = it & 1;                                                          \
    if (it + 1 < NIT) {                                                              \
      const __bf16* nA = gA + (it + 1) * 32;                                         \
      const __bf16* nB = gB + (it + 1) * 32;                                         \
      async_g2l_b128(&As[cur ^ 1][srow * LDT + scol],     nA);                       \
      async_g2l_b128(&As[cur ^ 1][srow * LDT + scol + 8], nA + 8);                   \
      async_g2l_b128(&Bs[cur ^ 1][srow * LDT + scol],     nB);                       \
      async_g2l_b128(&Bs[cur ^ 1][srow * LDT + scol + 8], nB + 8);                   \
    }                                                                                \
    if (it + 2 < NIT) {                                                              \
      __builtin_prefetch(gA + (it + 2) * 32, 0, 3);                                  \
      __builtin_prefetch(gB + (it + 2) * 32, 0, 3);                                  \
    }                                                                                \
    FragU af[4], bfr[2];                                                             \
    _Pragma("unroll")                                                                \
    for (int i = 0; i < 4; ++i) {                                                    \
      const __bf16* p = &As[cur][(wm * 64 + i * 16 + lrow) * LDT + lkA];             \
      af[i].q[0] = *(const uint4*)p;                                                 \
      af[i].q[1] = *(const uint4*)(p + 16);                                          \
    }                                                                                \
    _Pragma("unroll")                                                                \
    for (int j = 0; j < 2; ++j) {                                                    \
      const __bf16* p = &Bs[cur][(wn * 32 + j * 16 + lrow) * LDT + lkB];             \
      bfr[j].q[0] = *(const uint4*)p;                                                \
      bfr[j].q[1] = *(const uint4*)(p + 16);                                         \
    }                                                                                \
    _Pragma("unroll")                                                                \
    for (int i = 0; i < 4; ++i)                                                      \
      _Pragma("unroll")                                                              \
      for (int j = 0; j < 2; ++j)                                                    \
        acc[i][j] = __builtin_amdgcn_wmma_f32_16x16x32_bf16(                         \
            false, af[i].v, false, bfr[j].v, (short)0, acc[i][j], false, false);     \
    wait_async0();                                                                   \
    __syncthreads();                                                                 \
  }

// QKV projection: A=(4096,1024) bf16, W=(3072,1024) bf16, bias fp32.
// Epilogue scatters into per-head (B,H,L,64) bf16 Q/K/V; Q pre-scaled by
// (1/sqrt(64)) * log2(e) so the attention kernel can use exp2 directly.
__global__ __launch_bounds__(256) void qkv_gemm(
    const __bf16* __restrict__ A, const __bf16* __restrict__ W,
    const float* __restrict__ bias,
    __bf16* __restrict__ Qh, __bf16* __restrict__ Kh, __bf16* __restrict__ Vh) {
  GEMM_MAINLOOP(A, W, 1024)
  constexpr float QSCALE = 0.18033688011112042f;   // 0.125 * log2(e)
#pragma unroll
  for (int j = 0; j < 2; ++j) {
    const int n   = n0 + wn * 32 + j * 16 + lrow;
    const float bn = bias[n];
    const int t   = n >> 10;           // 0:Q 1:K 2:V
    const int rem = n & 1023;
    const int h   = rem >> 6;
    const int hd  = rem & 63;
    __bf16* dst = (t == 0) ? Qh : (t == 1) ? Kh : Vh;
    const float scl = (t == 0) ? QSCALE : 1.0f;
#pragma unroll
    for (int i = 0; i < 4; ++i) {
#pragma unroll
      for (int v = 0; v < 8; ++v) {
        const int m  = m0 + wm * 64 + i * 16 + lhi * 8 + v;
        const int bb = m >> 11;        // batch
        const int lr = m & 2047;       // sequence pos
        dst[(((size_t)(bb * CH + h)) * CL + lr) * CHD + hd] =
            f2bf((acc[i][j][v] + bn) * scl);
      }
    }
  }
}

// Output projection: A=attn(4096,1024) bf16, W=(1024,1024) bf16 -> fp32 out.
__global__ __launch_bounds__(256) void out_gemm(
    const __bf16* __restrict__ A, const __bf16* __restrict__ W,
    const float* __restrict__ bias, float* __restrict__ out) {
  GEMM_MAINLOOP(A, W, 1024)
#pragma unroll
  for (int j = 0; j < 2; ++j) {
    const int n = n0 + wn * 32 + j * 16 + lrow;
    const float bn = bias[n];
#pragma unroll
    for (int i = 0; i < 4; ++i) {
#pragma unroll
      for (int v = 0; v < 8; ++v) {
        const int m = m0 + wm * 64 + i * 16 + lhi * 8 + v;
        out[(size_t)m * CD + n] = acc[i][j][v] + bn;
      }
    }
  }
}

// ---------- fused causal flash attention (per (b,h)) ----------
// Block: 128 query rows (8 waves x 16 rows). Keys streamed in 32-row chunks:
// K tile arrives via double-buffered async DMA; V is transposed through VGPRs
// into a double-buffered LDS tile.
__global__ __launch_bounds__(256) void attn_fwd(
    const __bf16* __restrict__ Qh, const __bf16* __restrict__ Kh,
    const __bf16* __restrict__ Vh, __bf16* __restrict__ attn) {
  constexpr int KLD = 72;   // K tile row stride (elements), conflict-free
  constexpr int VLD = 40;   // V^T row stride
  constexpr int PLD = 40;   // per-wave P patch row stride
  __shared__ __align__(16) __bf16 Ks[2][32 * KLD];
  __shared__ __align__(16) __bf16 Vt[2][64 * VLD];
  __shared__ __align__(16) __bf16 Ps[8 * 16 * PLD];

  const int tid  = threadIdx.x;
  const int wave = tid >> 5, lane = tid & 31;
  const int lrow = lane & 15;
  const int lhi  = lane >> 4;
  const int lkA  = lhi << 3;
  const int lkB  = lhi << 3;
  const int bh = blockIdx.y;
  const int b  = bh >> 4, h = bh & 15;
  const __bf16* Qp = Qh + (size_t)bh * CL * CHD;
  const __bf16* Kp = Kh + (size_t)bh * CL * CHD;
  const __bf16* Vp = Vh + (size_t)bh * CL * CHD;
  const int q0 = blockIdx.x * 128 + wave * 16;

  // Q fragments for this wave's 16 rows (hd split into two K=32 chunks)
  FragU qf[2];
#pragma unroll
  for (int c = 0; c < 2; ++c) {
    const __bf16* p = Qp + (size_t)(q0 + lrow) * CHD + c * 32 + lkA;
    qf[c].q[0] = *(const uint4*)p;
    qf[c].q[1] = *(const uint4*)(p + 16);
  }

  f32x8 o[4] = {};
  float mrow[8], lsum[8];
#pragma unroll
  for (int v = 0; v < 8; ++v) { mrow[v] = -3.0e38f; lsum[v] = 0.0f; }

  const int sr = tid >> 3;            // staging row (key within chunk)
  const int sc = (tid & 7) << 3;      // staging col (hd)
  const int kend = blockIdx.x * 128 + 128;   // causal upper bound for this block
  const int nIt  = kend / 32;

  // prologue: stage chunk 0 into buffer 0
  async_g2l_b128(&Ks[0][sr * KLD + sc], Kp + (size_t)sr * CHD + sc);
  {
    uint4 vv = *(const uint4*)(Vp + (size_t)sr * CHD + sc);
    const __bf16* ve = (const __bf16*)&vv;
#pragma unroll
    for (int e = 0; e < 8; ++e) Vt[0][(sc + e) * VLD + sr] = ve[e];
  }
  wait_async0();
  __syncthreads();

  for (int it = 0; it < nIt; ++it) {
    const int kc  = it * 32;
    const int cur = it & 1;
    const bool has_next = (it + 1 < nIt);
    uint4 vv = {0, 0, 0, 0};
    if (has_next) {    // stream chunk it+1 while computing chunk it
      async_g2l_b128(&Ks[cur ^ 1][sr * KLD + sc],
                     Kp + (size_t)(kc + 32 + sr) * CHD + sc);
      vv = *(const uint4*)(Vp + (size_t)(kc + 32 + sr) * CHD + sc);
    }

    // S = Q @ K^T for two 16-key tiles (log2e*scale already folded into Q)
    f32x8 s0 = {}, s1 = {};
#pragma unroll
    for (int c = 0; c < 2; ++c) {
      FragU k0, k1;
      const __bf16* p0 = &Ks[cur][lrow * KLD + c * 32 + lkB];
      const __bf16* p1 = &Ks[cur][(16 + lrow) * KLD + c * 32 + lkB];
      k0.q[0] = *(const uint4*)p0; k0.q[1] = *(const uint4*)(p0 + 16);
      k1.q[0] = *(const uint4*)p1; k1.q[1] = *(const uint4*)(p1 + 16);
      s0 = __builtin_amdgcn_wmma_f32_16x16x32_bf16(false, qf[c].v, false, k0.v, (short)0, s0, false, false);
      s1 = __builtin_amdgcn_wmma_f32_16x16x32_bf16(false, qf[c].v, false, k1.v, (short)0, s1, false, false);
    }

    // causal mask + online softmax (row m lives in one 16-lane group)
    float p0a[8], p1a[8], alpha[8];
#pragma unroll
    for (int v = 0; v < 8; ++v) {
      const int qr = q0 + lhi * 8 + v;
      const float a = (kc + lrow      <= qr) ? s0[v] : -3.0e38f;
      const float c = (kc + 16 + lrow <= qr) ? s1[v] : -3.0e38f;
      const float mx   = redmax16(fmaxf(a, c));
      const float mnew = fmaxf(mrow[v], mx);
      alpha[v] = exp2f(mrow[v] - mnew);
      const float e0 = exp2f(a - mnew);
      const float e1 = exp2f(c - mnew);
      lsum[v] = lsum[v] * alpha[v] + redsum16(e0 + e1);
      mrow[v] = mnew;
      p0a[v] = e0; p1a[v] = e1;
    }
#pragma unroll
    for (int nt = 0; nt < 4; ++nt)
#pragma unroll
      for (int v = 0; v < 8; ++v)
        o[nt][v] = o[nt][v] * alpha[v];

    // C-layout -> A-layout via per-wave LDS patch (wave-local: dscnt wait only)
    __bf16* Pw = &Ps[wave * 16 * PLD];
#pragma unroll
    for (int v = 0; v < 8; ++v) {
      Pw[(lhi * 8 + v) * PLD + lrow]      = f2bf(p0a[v]);
      Pw[(lhi * 8 + v) * PLD + 16 + lrow] = f2bf(p1a[v]);
    }
    asm volatile("s_wait_dscnt 0" ::: "memory");   // CDNA5 split DS counter

    FragU pf;
    {
      const __bf16* p = &Pw[lrow * PLD + lkA];
      pf.q[0] = *(const uint4*)p;
      pf.q[1] = *(const uint4*)(p + 16);
    }
#pragma unroll
    for (int nt = 0; nt < 4; ++nt) {
      FragU vf;
      const __bf16* p = &Vt[cur][(nt * 16 + lrow) * VLD + lkB];
      vf.q[0] = *(const uint4*)p;
      vf.q[1] = *(const uint4*)(p + 16);
      o[nt] = __builtin_amdgcn_wmma_f32_16x16x32_bf16(false, pf.v, false, vf.v, (short)0, o[nt], false, false);
    }

    if (has_next) {   // finish staging V chunk it+1 (sinks below the compute)
      const __bf16* ve = (const __bf16*)&vv;
#pragma unroll
      for (int e = 0; e < 8; ++e) Vt[cur ^ 1][(sc + e) * VLD + sr] = ve[e];
    }
    wait_async0();
    __syncthreads();
  }

  // normalize + store concat-head layout (B, L, H*HD) as bf16 for out_gemm
#pragma unroll
  for (int v = 0; v < 8; ++v) lsum[v] = 1.0f / lsum[v];
#pragma unroll
  for (int nt = 0; nt < 4; ++nt) {
#pragma unroll
    for (int v = 0; v < 8; ++v) {
      const int row = q0 + lhi * 8 + v;
      attn[((size_t)b * CL + row) * CD + h * CHD + nt * 16 + lrow] =
          f2bf(o[nt][v] * lsum[v]);
    }
  }
}

// ---------- host launch ----------
extern "C" void kernel_launch(void* const* d_in, const int* in_sizes, int n_in,
                              void* d_out, int out_size, void* d_ws, size_t ws_size,
                              hipStream_t stream) {
  (void)in_sizes; (void)n_in; (void)out_size; (void)ws_size;
  const float* x     = (const float*)d_in[0];
  // d_in[1] = causal mask (bool) — implemented analytically, unused
  const float* W_qkv = (const float*)d_in[2];
  const float* b_qkv = (const float*)d_in[3];
  const float* W_out = (const float*)d_in[4];
  const float* b_out = (const float*)d_in[5];
  float* out = (float*)d_out;

  __bf16* ws = (__bf16*)d_ws;
  __bf16* xb    = ws; ws += (size_t)CM * CD;        // 8 MB
  __bf16* wqkvb = ws; ws += (size_t)3 * CD * CD;    // 6 MB
  __bf16* woutb = ws; ws += (size_t)CD * CD;        // 2 MB
  __bf16* Qh    = ws; ws += (size_t)CM * CD;        // 8 MB each
  __bf16* Kh    = ws; ws += (size_t)CM * CD;
  __bf16* Vh    = ws; ws += (size_t)CM * CD;
  __bf16* attnb = ws;                               // 8 MB  (total 48 MB)

  const int n_x = CM * CD, n_wq = 3 * CD * CD, n_wo = CD * CD;
  cvt_bf16<<<dim3((n_x  / 4 + 255) / 256), dim3(256), 0, stream>>>(x,     xb,    n_x);
  cvt_bf16<<<dim3((n_wq / 4 + 255) / 256), dim3(256), 0, stream>>>(W_qkv, wqkvb, n_wq);
  cvt_bf16<<<dim3((n_wo / 4 + 255) / 256), dim3(256), 0, stream>>>(W_out, woutb, n_wo);

  qkv_gemm<<<dim3(3 * CD / 128, CM / 128), dim3(256), 0, stream>>>(
      xb, wqkvb, b_qkv, Qh, Kh, Vh);

  attn_fwd<<<dim3(CL / 128, CB * CH), dim3(256), 0, stream>>>(Qh, Kh, Vh, attnb);

  out_gemm<<<dim3(CD / 128, CM / 128), dim3(256), 0, stream>>>(attnb, woutb, b_out, out);
}
